// GNNRNNModel_18270790877883
// MI455X (gfx1250) — matrix operand order repaired and checked by reading, not compiled, not run on hardware
//
#include <hip/hip_runtime.h>
#include <hip/hip_bf16.h>

#define NN  20000
#define EE  320000
#define TT  12
#define INC 128
#define GHD 64
#define RHD 128

typedef __attribute__((ext_vector_type(16))) __bf16 v16bf;
typedef __attribute__((ext_vector_type(8)))  float  v8f;

__device__ __forceinline__ float sigmoidf_(float x) {
    return 1.0f / (1.0f + __expf(-x));
}

// A-matrix 16x32 bf16 fragment (ISA 7.12.2): lane L holds row m=L&15;
// elements 0..7 are K = kb..kb+7, elements 8..15 are K = kb+16..kb+23,
// where kb = (L>>4)*8 (plus the caller's K-chunk offset).
__device__ __forceinline__ v16bf a_frag_from_f32(const float* p) {
    v16bf v;
#pragma unroll
    for (int i = 0; i < 8; ++i) v[i] = (__bf16)p[i];
#pragma unroll
    for (int i = 0; i < 8; ++i) v[8 + i] = (__bf16)p[16 + i];
    return v;
}

__device__ __forceinline__ v16bf a_frag_from_lds(const __bf16* p) {
    v16bf v;
#pragma unroll
    for (int i = 0; i < 8; ++i) v[i] = p[i];
#pragma unroll
    for (int i = 0; i < 8; ++i) v[8 + i] = p[16 + i];
    return v;
}

// ---------------------------------------------------------------- degree/norm
__global__ void deg_init_kernel(float* __restrict__ deg) {
    int n = blockIdx.x * blockDim.x + threadIdx.x;
    if (n < NN) deg[n] = 1.0f;                 // self loop
}

__global__ void deg_edge_kernel(const int* __restrict__ ei, float* __restrict__ deg) {
    int e = blockIdx.x * blockDim.x + threadIdx.x;
    if (e < EE) unsafeAtomicAdd(&deg[ei[EE + e]], 1.0f);
}

__global__ void dinv_kernel(const float* __restrict__ deg, float* __restrict__ dinv) {
    int n = blockIdx.x * blockDim.x + threadIdx.x;
    if (n < NN) dinv[n] = __frsqrt_rn(deg[n]);
}

// ------------------------------------------------- GCN transform (bf16 WMMA)
// hraw[T*N, 64] = bf16(x[T*N,128]) @ bf16(gcn_w[128,64]), f32 accumulate.
__global__ void __launch_bounds__(256, 1)
gcn_matmul_kernel(const float* __restrict__ x,
                  const float* __restrict__ gw,
                  float* __restrict__ hraw) {
    __shared__ __bf16 WG[4 * 4 * 32 * 16];     // [nt][kc][lane][e], 16 KB
    const int tid = threadIdx.x;

    // Stage B fragments: B[k][n] = gcn_w[k*64+n]; lane holds col n=L&15,
    // elements e -> K = kc*32 + ((L>>4)<<3) + e + (e&8) (ISA table order, 32B per lane).
    for (int p = tid; p < 4 * 4 * 32; p += 256) {
        int lane = p & 31;
        int frag = p >> 5;                     // nt*4 + kc
        int kc = frag & 3, nt = frag >> 2;
        int n  = nt * 16 + (lane & 15);
        int kb = kc * 32 + ((lane >> 4) << 3);
        __bf16* dp = WG + frag * 512 + lane * 16;
#pragma unroll
        for (int e = 0; e < 16; ++e) dp[e] = (__bf16)gw[(kb + e + (e & 8)) * GHD + n];
    }
    __syncthreads();

    const int lane = tid & 31, wv = tid >> 5;
    const int rowbase = blockIdx.x * 128 + wv * 16;   // 240000/128 = 1875 exact
    const int arow = rowbase + (lane & 15);
    const int kb8  = (lane >> 4) << 3;

    v16bf afrag[4];
#pragma unroll
    for (int kc = 0; kc < 4; ++kc)
        afrag[kc] = a_frag_from_f32(x + (size_t)arow * INC + kc * 32 + kb8);

    // Preload all 16 B fragments, then issue 16 back-to-back WMMAs.
    v16bf bfrag[4][4];
#pragma unroll
    for (int nt = 0; nt < 4; ++nt)
#pragma unroll
        for (int kc = 0; kc < 4; ++kc)
            bfrag[nt][kc] = *(const v16bf*)(WG + (nt * 4 + kc) * 512 + lane * 16);

    const int nidx  = lane & 15;
    const int mbase = (lane >> 4) << 3;
#pragma unroll
    for (int nt = 0; nt < 4; ++nt) {
        v8f acc = {};
#pragma unroll
        for (int kc = 0; kc < 4; ++kc)
            acc = __builtin_amdgcn_wmma_f32_16x16x32_bf16(
                false, afrag[kc], false, bfrag[nt][kc], (short)0, acc, false, false);
#pragma unroll
        for (int r = 0; r < 8; ++r)
            hraw[(size_t)(rowbase + mbase + r) * GHD + nt * 16 + nidx] = acc[r];
    }
}

// ---------------------------------------- GCN aggregate: self-loop + bias init
__global__ void hs_init_kernel(const float* __restrict__ hraw,
                               const float* __restrict__ dinv,
                               const float* __restrict__ gb,
                               float* __restrict__ hs) {
    int idx = blockIdx.x * blockDim.x + threadIdx.x;     // float4 units
    if (idx >= TT * NN * (GHD / 4)) return;
    int ch4 = idx & 15;
    int row = idx >> 4;
    int n   = row % NN;
    float s = dinv[n]; s = s * s;
    float4 v = ((const float4*)hraw)[idx];
    float4 b = ((const float4*)gb)[ch4];
    float4 o = { v.x * s + b.x, v.y * s + b.y, v.z * s + b.z, v.w * s + b.w };
    ((float4*)hs)[idx] = o;
}

// ------------------------------ GCN aggregate: edge scatter (L2-resident f32)
__global__ void edge_scatter_kernel(const int* __restrict__ ei,
                                    const float* __restrict__ dinv,
                                    const float* __restrict__ hraw,
                                    float* __restrict__ hs) {
    int idx = blockIdx.x * blockDim.x + threadIdx.x;     // (t, e)
    if (idx >= TT * EE) return;
    int t = idx / EE;
    int e = idx - t * EE;
    int s = ei[e];
    int d = ei[EE + e];
    float nrm = dinv[s] * dinv[d];
    const float4* sp = (const float4*)(hraw + ((size_t)t * NN + s) * GHD);
    float* dp = hs + ((size_t)t * NN + d) * GHD;
#pragma unroll 4
    for (int q = 0; q < GHD / 4; ++q) {
        float4 v = sp[q];
        unsafeAtomicAdd(dp + q * 4 + 0, nrm * v.x);
        unsafeAtomicAdd(dp + q * 4 + 1, nrm * v.y);
        unsafeAtomicAdd(dp + q * 4 + 2, nrm * v.z);
        unsafeAtomicAdd(dp + q * 4 + 3, nrm * v.w);
    }
}

// ------------------------------------------- fused LSTM over all T timesteps
// Per block: 128 nodes, 8 waves x 16 nodes. Weights [192][512] fragment-ordered
// bf16 in LDS (192 KB). h state bf16 in LDS, c state fp32 in registers laid out
// in the C/D tile pattern. gates = [x_t | h] @ Wcat, K = 64 + 128 = 192.
__global__ void __launch_bounds__(256, 1)
lstm_kernel(const float* __restrict__ hs,
            const float* __restrict__ w_ih,
            const float* __restrict__ w_hh,
            const float* __restrict__ b_ih,
            const float* __restrict__ b_hh,
            float* __restrict__ hlast) {
    extern __shared__ __align__(16) char smem_raw[];
    __bf16* WL = (__bf16*)smem_raw;             // 8*4*6*512 = 98304 elems
    __bf16* HS = WL + 8 * 4 * 6 * 512;          // 128*128   = 16384 elems

    const int tid = threadIdx.x;

    // Stage weights fragment-ordered: frag = (c*4+g)*6+kc,
    // B[k][n] with n = g*128 + c*16 + (L&15), K = kc*32 + ((L>>4)<<3) + e + (e&8).
    for (int p = tid; p < 8 * 4 * 6 * 32; p += 256) {
        int lane = p & 31;
        int frag = p >> 5;
        int kc = frag % 6;
        int cg = frag / 6;
        int g = cg & 3, c = cg >> 2;
        int n  = g * RHD + c * 16 + (lane & 15);
        int kb = kc * 32 + ((lane >> 4) << 3);
        __bf16* dp = WL + frag * 512 + lane * 16;
#pragma unroll
        for (int e = 0; e < 16; ++e) {
            int k = kb + e + (e & 8);
            float w = (k < GHD) ? w_ih[(size_t)n * GHD + k]
                                : w_hh[(size_t)n * RHD + (k - GHD)];
            dp[e] = (__bf16)w;
        }
    }
    for (int i = tid; i < 128 * RHD; i += 256) HS[i] = (__bf16)0.0f;
    __syncthreads();

    const int lane = tid & 31, wv = tid >> 5;
    const int lrow    = wv * 16 + (lane & 15);            // local A row
    const int gnode_a = blockIdx.x * 128 + lrow;
    const int arow    = (gnode_a < NN) ? gnode_a : (NN - 1);
    const int kb8     = (lane >> 4) << 3;
    const int mb      = (lane >> 4) << 3;                 // C/D row base
    const int nn      = lane & 15;                        // C/D col

    float bias[8][4];
#pragma unroll
    for (int c = 0; c < 8; ++c)
#pragma unroll
        for (int g = 0; g < 4; ++g) {
            int n = g * RHD + c * 16 + nn;
            bias[c][g] = b_ih[n] + b_hh[n];
        }

    float cst[8][8];
#pragma unroll
    for (int c = 0; c < 8; ++c)
#pragma unroll
        for (int r = 0; r < 8; ++r) cst[c][r] = 0.0f;

    for (int t = 0; t < TT; ++t) {
        v16bf afrag[6];
        const float* xr = hs + ((size_t)t * NN + arow) * GHD;
#pragma unroll
        for (int kc = 0; kc < 2; ++kc)                    // K 0..63 from x_t
            afrag[kc] = a_frag_from_f32(xr + kc * 32 + kb8);
#pragma unroll
        for (int kc = 0; kc < 4; ++kc)                    // K 64..191 from h
            afrag[2 + kc] = a_frag_from_lds(HS + lrow * RHD + kc * 32 + kb8);

        // Two col-chunks in flight: 8 independent accumulator tiles so the
        // scheduler can hoist ds_loads of B across WMMAs.
#pragma unroll 2
        for (int c = 0; c < 8; ++c) {
            v8f acc[4];
#pragma unroll
            for (int g = 0; g < 4; ++g) {
                v8f a = {};
#pragma unroll
                for (int kc = 0; kc < 6; ++kc) {
                    const v16bf b = *(const v16bf*)(
                        WL + (((c * 4 + g) * 6 + kc) << 9) + (lane << 4));
                    a = __builtin_amdgcn_wmma_f32_16x16x32_bf16(
                        false, afrag[kc], false, b, (short)0, a, false, false);
                }
                acc[g] = a;
            }
#pragma unroll
            for (int r = 0; r < 8; ++r) {
                float ig = sigmoidf_(acc[0][r] + bias[c][0]);
                float fg = sigmoidf_(acc[1][r] + bias[c][1]);
                float gg = tanhf(acc[2][r] + bias[c][2]);
                float og = sigmoidf_(acc[3][r] + bias[c][3]);
                float cn = fg * cst[c][r] + ig * gg;
                cst[c][r] = cn;
                float hn = og * tanhf(cn);
                int lr = wv * 16 + mb + r;
                HS[lr * RHD + c * 16 + nn] = (__bf16)hn;
                if (t == TT - 1) {
                    int gn = blockIdx.x * 128 + lr;
                    if (gn < NN)
                        hlast[(size_t)gn * RHD + c * 16 + nn] = hn;
                }
            }
        }
    }
}

// ----------------------------------------------------------------- final FC
__global__ void fc_kernel(const float* __restrict__ hlast,
                          const float* __restrict__ fw,
                          const float* __restrict__ fb,
                          float* __restrict__ out) {
    int n = blockIdx.x * blockDim.x + threadIdx.x;
    if (n >= NN) return;
    const float4* h4 = (const float4*)(hlast + (size_t)n * RHD);
    const float4* w4 = (const float4*)fw;
    float acc = 0.0f;
#pragma unroll
    for (int q = 0; q < RHD / 4; ++q) {
        float4 h = h4[q], w = w4[q];
        acc += h.x * w.x + h.y * w.y + h.z * w.z + h.w * w.w;
    }
    out[n] = acc + fb[0];
}

extern "C" void kernel_launch(void* const* d_in, const int* in_sizes, int n_in,
                              void* d_out, int out_size, void* d_ws, size_t ws_size,
                              hipStream_t stream) {
    (void)in_sizes; (void)n_in; (void)out_size; (void)ws_size;
    const float* x_seq = (const float*)d_in[0];
    const int*   ei    = (const int*)d_in[1];
    const float* gcn_w = (const float*)d_in[2];
    const float* gcn_b = (const float*)d_in[3];
    const float* w_ih  = (const float*)d_in[4];
    const float* w_hh  = (const float*)d_in[5];
    const float* b_ih  = (const float*)d_in[6];
    const float* b_hh  = (const float*)d_in[7];
    const float* fc_w  = (const float*)d_in[8];
    const float* fc_b  = (const float*)d_in[9];
    float* out = (float*)d_out;

    float* ws    = (float*)d_ws;
    float* deg   = ws;                                   // N
    float* dinv  = deg + NN;                             // N
    float* hraw  = dinv + NN;                            // T*N*64
    float* hsbuf = hraw + (size_t)TT * NN * GHD;         // T*N*64
    float* hlast = hsbuf + (size_t)TT * NN * GHD;        // N*128

    deg_init_kernel<<<(NN + 255) / 256, 256, 0, stream>>>(deg);
    deg_edge_kernel<<<(EE + 255) / 256, 256, 0, stream>>>(ei, deg);
    dinv_kernel<<<(NN + 255) / 256, 256, 0, stream>>>(deg, dinv);

    gcn_matmul_kernel<<<(TT * NN) / 128, 256, 0, stream>>>(x_seq, gcn_w, hraw);

    hs_init_kernel<<<(TT * NN * (GHD / 4) + 255) / 256, 256, 0, stream>>>(
        hraw, dinv, gcn_b, hsbuf);
    edge_scatter_kernel<<<(TT * EE + 255) / 256, 256, 0, stream>>>(
        ei, dinv, hraw, hsbuf);

    const size_t lstm_smem = (size_t)(8 * 4 * 6 * 512 + 128 * RHD) * sizeof(__bf16);
    lstm_kernel<<<(NN + 127) / 128, 256, lstm_smem, stream>>>(
        hsbuf, w_ih, w_hh, b_ih, b_hh, hlast);

    fc_kernel<<<(NN + 255) / 256, 256, 0, stream>>>(hlast, fc_w, fc_b, out);
}
